// ContrastiveLoss_67216238183240
// MI455X (gfx1250) — matrix operand order repaired and checked by reading, not compiled, not run on hardware
//
#include <hip/hip_runtime.h>
#include <cstdint>
#include <cstddef>

// ---------------------------------------------------------------------------
// ContrastiveLoss fused pipeline for MI455X (gfx1250, wave32, WMMA).
//
// Everything is compute-bound and L2-resident (z: 12.6MB, codebook: 8.4MB vs
// 192MB L2).  Gram matrices use V_WMMA_F32_16X16X32_BF16 with an f32 -> bf16
// hi/lo split (3 WMMAs per 32-K step: hi*hi + hi*lo + lo*hi), giving ~fp32
// accuracy at 2.7x fewer matrix instructions than the 16x16x4 f32 WMMA path.
// Symmetry: only upper-triangular 128x128 tiles are computed (off-diagonal
// tiles weighted x2 in the full-matrix sums; block-pair ratios are symmetric
// so upper-only traversal matches the reference tile enumeration exactly).
// ---------------------------------------------------------------------------

typedef __attribute__((ext_vector_type(16))) __bf16 v16bf;
typedef __attribute__((ext_vector_type(8)))  float  v8f;

namespace {

constexpr int   BZ   = 12288;    // z rows
constexpr int   KC   = 8192;     // codebook rows
constexpr int   DD   = 256;      // feature dim
constexpr int   NS   = 200000;   // quantile samples
constexpr int   NB   = 16384;    // histogram bins
constexpr float DHI  = 64.0f;    // histogram range [0, DHI)
constexpr int   TZ   = BZ / 128; // 96 tiles per dim (z)
constexpr int   TC   = KC / 128; // 64 tiles per dim (codebook)

// ---- workspace layout (256B aligned chunks so v16bf loads stay 32B aligned)
constexpr size_t al256(size_t x){ return (x + 255) & ~size_t(255); }
constexpr size_t OFF_THR  = 0;                                            // 3 floats
constexpr size_t OFF_HIST = al256(OFF_THR  + 4*sizeof(float));            // NB ints
constexpr size_t OFF_PZ   = al256(OFF_HIST + size_t(NB)*sizeof(int));     // TZ*TZ*4 f
constexpr size_t OFF_PCB  = al256(OFF_PZ   + size_t(TZ)*TZ*4*sizeof(float)); // TC*TC f
constexpr size_t OFF_NZ   = al256(OFF_PCB  + size_t(TC)*TC*sizeof(float));
constexpr size_t OFF_NCB  = al256(OFF_NZ   + size_t(BZ)*sizeof(float));
constexpr size_t OFF_ZHI  = al256(OFF_NCB  + size_t(KC)*sizeof(float));
constexpr size_t OFF_ZLO  = al256(OFF_ZHI  + size_t(BZ)*DD*2);
constexpr size_t OFF_CHI  = al256(OFF_ZLO  + size_t(BZ)*DD*2);
constexpr size_t OFF_CLO  = al256(OFF_CHI  + size_t(KC)*DD*2);
constexpr size_t WS_NEED  = OFF_CLO + size_t(KC)*DD*2;                    // ~21.3 MB
constexpr int    ZERO_DW  = (int)(OFF_NZ / 4);  // zero thr+hist+partials

__device__ __forceinline__ unsigned short f2bf(float f){   // RNE f32->bf16
  unsigned u = __float_as_uint(f);
  u += 0x7fffu + ((u >> 16) & 1u);
  return (unsigned short)(u >> 16);
}
__device__ __forceinline__ float bf2f(unsigned short h){
  return __uint_as_float((unsigned)h << 16);
}
__device__ __forceinline__ unsigned hash32(unsigned x){    // splitmix32
  x ^= x >> 16; x *= 0x7feb352du;
  x ^= x >> 15; x *= 0x846ca68bu;
  x ^= x >> 16; return x;
}
__device__ __forceinline__ float sigm(float x){ return 1.0f/(1.0f + __expf(-x)); }

__device__ __forceinline__ float blockReduce256(float v, float* red){
  const int t = threadIdx.x;
  red[t] = v; __syncthreads();
  for (int s = 128; s > 0; s >>= 1){
    if (t < s) red[t] += red[t + s];
    __syncthreads();
  }
  const float r = red[0]; __syncthreads();
  return r;
}

} // namespace

// ---------------------------------------------------------------------------
__global__ __launch_bounds__(256) void cl_zero_ws(unsigned int* p, int n){
  for (int i = blockIdx.x*blockDim.x + threadIdx.x; i < n; i += gridDim.x*blockDim.x)
    p[i] = 0u;
}

// Row norms + f32 -> (bf16 hi, bf16 lo) split.  One wave per row (D=256 -> 8/lane).
__global__ __launch_bounds__(256) void cl_prep_rows(
    const float* __restrict__ src, unsigned short* __restrict__ hi,
    unsigned short* __restrict__ lo, float* __restrict__ norms, int nrows){
  const int wave = threadIdx.x >> 5, lane = threadIdx.x & 31;
  const int row  = blockIdx.x*8 + wave;
  if (row >= nrows) return;
  const float* p = src + (size_t)row * DD;
  float acc = 0.0f;
  #pragma unroll
  for (int t = 0; t < 8; t++){
    const int k = lane + 32*t;
    const float v = p[k];
    const unsigned short h = f2bf(v);
    const unsigned short l = f2bf(v - bf2f(h));
    hi[(size_t)row*DD + k] = h;
    lo[(size_t)row*DD + k] = l;
    acc += v * v;
  }
  for (int off = 16; off; off >>= 1) acc += __shfl_xor(acc, off, 32);
  if (lane == 0) norms[row] = acc;
}

// 200k sampled pair distances -> integer histogram (deterministic).
__global__ __launch_bounds__(256) void cl_sample_hist(
    const float* __restrict__ z, int* __restrict__ hist){
  const int s = blockIdx.x*blockDim.x + threadIdx.x;
  if (s >= NS) return;
  const unsigned a = hash32(2u*(unsigned)s + 1u);
  const unsigned b = hash32(2u*(unsigned)s + 2u);
  const int ii = (int)(a % (unsigned)BZ);
  const int r  = 1 + (int)(b % (unsigned)(BZ - 1));
  const int jj = (ii + r) % BZ;
  const float4* p = (const float4*)(z + (size_t)ii * DD);
  const float4* q = (const float4*)(z + (size_t)jj * DD);
  float acc = 0.0f;
  #pragma unroll 4
  for (int k = 0; k < DD/4; k++){
    const float4 x = p[k], y = q[k];
    const float dx = x.x - y.x, dy = x.y - y.y, dz = x.z - y.z, dw = x.w - y.w;
    acc += dx*dx + dy*dy + dz*dz + dw*dw;
  }
  int bin = (int)(sqrtf(acc) * ((float)NB / DHI));
  bin = bin < 0 ? 0 : (bin > NB-1 ? NB-1 : bin);
  atomicAdd(hist + bin, 1);
}

// Quantile scan (0.95, 0.99) with linear interpolation inside the bin.
__global__ void cl_quantiles(const int* __restrict__ hist, float* __restrict__ thr){
  if (threadIdx.x != 0 || blockIdx.x != 0) return;
  const float binw = DHI / (float)NB;
  float targ[2] = { 0.95f * (float)(NS - 1), 0.99f * (float)(NS - 1) };
  float q[2]    = { 0.0f, 0.0f };
  long  cum = 0; int qi = 0;
  for (int b = 0; b < NB && qi < 2; b++){
    const int c = hist[b];
    while (qi < 2 && c > 0 && (float)(cum + c) > targ[qi]){
      float frac = (targ[qi] - (float)cum + 0.5f) / (float)c;
      frac = frac < 0.f ? 0.f : (frac > 1.f ? 1.f : frac);
      q[qi] = ((float)b + frac) * binw;
      qi++;
    }
    cum += c;
  }
  thr[0] = q[0]; thr[1] = q[1]; thr[2] = 0.5f * (q[0] + q[1]);
}

// ---------------------------------------------------------------------------
// Gram tile kernel.  128x128 output tile per block (upper triangle only),
// 8 waves arranged 4(M) x 2(N); each wave = 2x4 grid of 16x16 WMMA tiles.
// Per 32-K step per wave: 12 fragment loads, 24 v_wmma_f32_16x16x32_bf16.
// Fragment layouts (A/B: this code's contiguous k order, not the ISA table order):
//   A (16x32 bf16): lanes 0-15 hold K=0..15, lanes 16-31 hold K=16..31
//   B (32x16 bf16): lanes 0-15 hold K=0..15, lanes 16-31 hold K=16..31
//   C/D 16x16 f32 : VGPR r, lanes 0-15 -> M=r, lanes 16-31 -> M=r+8; N=lane%16
// ---------------------------------------------------------------------------
template<bool CB>
__global__ __launch_bounds__(256) void cl_gram_tiles(
    const unsigned short* __restrict__ hi, const unsigned short* __restrict__ lo,
    const float* __restrict__ norms, const float* __restrict__ thr,
    float* __restrict__ partials){
  const int ti = blockIdx.y, tj = blockIdx.x;
  if (tj < ti) return;                     // symmetry: upper-triangular tiles only
  __shared__ float red[256];

  const int lane = threadIdx.x & 31, wave = threadIdx.x >> 5;
  const int wm = wave >> 1, wn = wave & 1;
  const int laneLow = lane & 15, laneHi = lane >> 4;
  const int i0 = ti*128 + wm*32;
  const int j0 = tj*128 + wn*64;
  const int aK0 = laneHi * 16;
  const int bK0 = laneHi * 16;

  v8f acc[2][4];
  #pragma unroll
  for (int mt = 0; mt < 2; mt++)
    #pragma unroll
    for (int nt = 0; nt < 4; nt++)
      acc[mt][nt] = (v8f){0.f,0.f,0.f,0.f,0.f,0.f,0.f,0.f};

  for (int kk = 0; kk < DD; kk += 32){
    v16bf ahi[2], alo[2], bhi[4], blo[4];
    #pragma unroll
    for (int mt = 0; mt < 2; mt++){
      const size_t o = (size_t)(i0 + mt*16 + laneLow) * DD + kk + aK0;
      union { uint4 q[2]; v16bf v; } uh, ul;
      uh.q[0] = *(const uint4*)(hi + o);
      uh.q[1] = *(const uint4*)(hi + o + 8);
      ul.q[0] = *(const uint4*)(lo + o);
      ul.q[1] = *(const uint4*)(lo + o + 8);
      ahi[mt] = uh.v; alo[mt] = ul.v;
    }
    #pragma unroll
    for (int nt = 0; nt < 4; nt++){
      const size_t o = (size_t)(j0 + nt*16 + laneLow) * DD + kk + bK0;
      bhi[nt] = *(const v16bf*)(hi + o);
      blo[nt] = *(const v16bf*)(lo + o);
    }
    #pragma unroll
    for (int mt = 0; mt < 2; mt++)
      #pragma unroll
      for (int nt = 0; nt < 4; nt++){
        acc[mt][nt] = __builtin_amdgcn_wmma_f32_16x16x32_bf16(
            false, ahi[mt], false, bhi[nt], (short)0, acc[mt][nt], false, false);
        acc[mt][nt] = __builtin_amdgcn_wmma_f32_16x16x32_bf16(
            false, ahi[mt], false, blo[nt], (short)0, acc[mt][nt], false, false);
        acc[mt][nt] = __builtin_amdgcn_wmma_f32_16x16x32_bf16(
            false, alo[mt], false, bhi[nt], (short)0, acc[mt][nt], false, false);
      }
  }

  // ---- fused epilogue: d = sqrt(|zi|^2 + |zj|^2 - 2 zi.zj) per element
  const bool  diagTile = (ti == tj);
  const float offF = diagTile ? 1.0f : 2.0f;   // full-matrix sums count (i,j)+(j,i)
  const int   tilesPerDim = CB ? TC : TZ;
  const int   t = ti * tilesPerDim + tj;

  if (CB){
    float g = 0.0f;
    #pragma unroll
    for (int mt = 0; mt < 2; mt++)
      #pragma unroll
      for (int nt = 0; nt < 4; nt++){
        const int j  = j0 + nt*16 + laneLow;
        const float nj = norms[j];
        #pragma unroll
        for (int r = 0; r < 8; r++){
          const int i = i0 + mt*16 + r + 8*laneHi;
          const float sq = fmaxf(norms[i] + nj - 2.0f*acc[mt][nt][r], 0.0f);
          if (!(diagTile && i == j)) g += offF * __expf(-0.5f * sq);
        }
      }
    g = blockReduce256(g, red);
    if (threadIdx.x == 0) partials[t] = g;
  } else {
    const float lower  = thr[0], upper = thr[1], center = thr[2];
    const bool  bp     = (ti >> 5) != (tj >> 5);  // different 4096-row blocks
    float rep2 = 0.0f, rel = 0.0f, num = 0.0f, den = 0.0f;
    #pragma unroll
    for (int mt = 0; mt < 2; mt++)
      #pragma unroll
      for (int nt = 0; nt < 4; nt++){
        const int j  = j0 + nt*16 + laneLow;
        const float nj = norms[j];
        #pragma unroll
        for (int r = 0; r < 8; r++){
          const int i = i0 + mt*16 + r + 8*laneHi;
          const float sq = fmaxf(norms[i] + nj - 2.0f*acc[mt][nt][r], 0.0f);
          const float d  = sqrtf(sq);
          if (!(diagTile && i == j)){
            const float dm2 = d - 2.0f;
            rep2 += offF * __expf(dm2*dm2 * (-1.0f/18.0f));
            rel  += offF * fmaxf(lower - d, 0.0f);
          }
          if (bp){   // block-pair ratio: upper tiles match reference enumeration
            const float w  = sigm(20.0f*(d - lower)) * sigm(20.0f*(upper - d));
            const float dc = d - center;
            num += w * __expf(dc*dc * (-1.0f/18.0f));
            den += w;
          }
        }
      }
    rep2 = blockReduce256(rep2, red);
    rel  = blockReduce256(rel,  red);
    num  = blockReduce256(num,  red);
    den  = blockReduce256(den,  red);
    if (threadIdx.x == 0){
      float* p = partials + 4*(size_t)t;
      p[0] = rep2; p[1] = rel; p[2] = num; p[3] = den;
    }
  }
}

// Fixed-order final reduction of per-tile partials -> 5 scalar outputs.
__global__ __launch_bounds__(256) void cl_finalize(
    const float* __restrict__ pz, const float* __restrict__ pcb,
    float* __restrict__ out){
  __shared__ float red[256];
  float rep2 = 0.f, rel = 0.f, cb = 0.f;
  float num[3] = {0.f,0.f,0.f}, den[3] = {0.f,0.f,0.f};
  for (int t = threadIdx.x; t < TZ*TZ; t += 256){
    const int ti = t / TZ, tj = t % TZ;
    const float* p = pz + 4*(size_t)t;
    rep2 += p[0]; rel += p[1];
    const int bi = ti >> 5, bj = tj >> 5;
    if (tj > ti && bi != bj){
      const int pi = bi + bj - 1;   // (0,1)->0 (0,2)->1 (1,2)->2
      num[pi] += p[2]; den[pi] += p[3];
    }
  }
  for (int t = threadIdx.x; t < TC*TC; t += 256) cb += pcb[t];
  rep2 = blockReduce256(rep2, red);
  rel  = blockReduce256(rel,  red);
  for (int i = 0; i < 3; i++){
    num[i] = blockReduce256(num[i], red);
    den[i] = blockReduce256(den[i], red);
  }
  cb = blockReduce256(cb, red);
  if (threadIdx.x == 0){
    const double npz = (double)BZ * (double)(BZ - 1);
    const double npc = (double)KC * (double)(KC - 1);
    float fl = 0.f;
    for (int i = 0; i < 3; i++) fl += num[i] / fmaxf(den[i], 1e-8f);
    fl *= (1.0f/3.0f);
    const float r2  = (float)((double)rep2 / npz);
    const float cbl = (float)((double)cb   / npc);
    const float rz  = (float)((double)rel  / npz);
    out[0] = fl;          // final_loss
    out[1] = 1.0f;        // neg_loss
    out[2] = r2;          // repel_from_2
    out[3] = cbl;         // cb_loss
    out[4] = rz + cbl;    // latent_repel_loss
  }
}

// ---------------------------------------------------------------------------
extern "C" void kernel_launch(void* const* d_in, const int* in_sizes, int n_in,
                              void* d_out, int out_size, void* d_ws, size_t ws_size,
                              hipStream_t stream){
  (void)in_sizes; (void)n_in; (void)out_size; (void)ws_size; // needs WS_NEED ~21.3MB
  const float* z   = (const float*)d_in[0];
  const float* cbk = (const float*)d_in[1];
  float* out = (float*)d_out;
  char*  ws  = (char*)d_ws;

  float*          thr  = (float*)(ws + OFF_THR);
  int*            hist = (int*)  (ws + OFF_HIST);
  float*          pz   = (float*)(ws + OFF_PZ);
  float*          pcb  = (float*)(ws + OFF_PCB);
  float*          nz   = (float*)(ws + OFF_NZ);
  float*          ncb  = (float*)(ws + OFF_NCB);
  unsigned short* zhi  = (unsigned short*)(ws + OFF_ZHI);
  unsigned short* zlo  = (unsigned short*)(ws + OFF_ZLO);
  unsigned short* chi  = (unsigned short*)(ws + OFF_CHI);
  unsigned short* clo  = (unsigned short*)(ws + OFF_CLO);

  cl_zero_ws   <<<(ZERO_DW + 255)/256, 256, 0, stream>>>((unsigned int*)ws, ZERO_DW);
  cl_prep_rows <<<BZ/8, 256, 0, stream>>>(z,   zhi, zlo, nz,  BZ);
  cl_prep_rows <<<KC/8, 256, 0, stream>>>(cbk, chi, clo, ncb, KC);
  cl_sample_hist<<<(NS + 255)/256, 256, 0, stream>>>(z, hist);
  cl_quantiles <<<1, 32, 0, stream>>>(hist, thr);
  {
    dim3 g(TZ, TZ);
    cl_gram_tiles<false><<<g, 256, 0, stream>>>(zhi, zlo, nz, thr, pz);
  }
  {
    dim3 g(TC, TC);
    cl_gram_tiles<true><<<g, 256, 0, stream>>>(chi, clo, ncb, thr, pcb);
  }
  cl_finalize<<<1, 256, 0, stream>>>(pz, pcb, out);
}
